// SVDQW4A4Linear_37452114821821
// MI455X (gfx1250) — 8-bit matrix operand order repaired and checked by reading, compiled, not run on hardware
//
#include <hip/hip_runtime.h>
#include <hip/hip_bf16.h>
#include <stdint.h>

#define M_DIM 256
#define C_DIM 8192
#define O_DIM 8192
#define NG    128
#define RANK  32

typedef __attribute__((ext_vector_type(8))) int v8i;

__device__ __forceinline__ int pack4(int a, int b, int c, int d) {
    return (a & 0xFF) | ((b & 0xFF) << 8) | ((c & 0xFF) << 16) | ((d & 0xFF) << 24);
}

__global__ __launch_bounds__(256) void quant_act_kernel(
    const float* __restrict__ x,
    const float* __restrict__ smooth,
    int8_t* __restrict__ qact,
    float* __restrict__ ascales)
{
    const int gw   = (blockIdx.x * 256 + threadIdx.x) >> 5;
    const int lane = threadIdx.x & 31;
    const int m = gw >> 7;
    const int g = gw & (NG - 1);
    const int c = g * 64 + lane * 2;

    const size_t base = (size_t)m * C_DIM + c;
    float v0 = x[base + 0] / smooth[c + 0];
    float v1 = x[base + 1] / smooth[c + 1];

    float amax = fmaxf(fabsf(v0), fabsf(v1));
    #pragma unroll
    for (int off = 16; off > 0; off >>= 1)
        amax = fmaxf(amax, __shfl_xor(amax, off, 32));

    float sc = fmaxf(amax * (1.0f / 7.0f), 1e-8f);
    float rs = 1.0f / sc;
    int q0 = (int)rintf(v0 * rs); q0 = min(7, max(-8, q0));
    int q1 = (int)rintf(v1 * rs); q1 = min(7, max(-8, q1));

    uint16_t packed = (uint16_t)((q0 & 0xFF) | ((q1 & 0xFF) << 8));
    *(uint16_t*)(qact + base) = packed;

    if (lane == 0) ascales[(size_t)m * NG + g] = sc;
}

__global__ __launch_bounds__(256) void lora_down_kernel(
    const float* __restrict__ x,
    const float* __restrict__ pd,
    float* __restrict__ tmp)
{
    const int m     = blockIdx.x;
    const int lane  = threadIdx.x & 31;
    const int chunk = threadIdx.x >> 5;

    const float* xr  = x  + (size_t)m * C_DIM + chunk * 1024;
    const float* pdp = pd + (size_t)chunk * 1024 * RANK + lane;

    float acc = 0.0f;
    #pragma unroll 4
    for (int c = 0; c < 1024; ++c)
        acc = fmaf(xr[c], pdp[(size_t)c * RANK], acc);

    __shared__ float red[256];
    red[threadIdx.x] = acc;
    __syncthreads();
    if (threadIdx.x < 32) {
        float s = 0.0f;
        #pragma unroll
        for (int ch = 0; ch < 8; ++ch) s += red[ch * 32 + threadIdx.x];
        tmp[(size_t)m * RANK + threadIdx.x] = s;
    }
}

__global__ __launch_bounds__(256) void lora_up_bias_kernel(
    const float* __restrict__ tmp,
    const float* __restrict__ pu,
    const float* __restrict__ bias,
    float* __restrict__ out)
{
    const int n = blockIdx.x * 256 + threadIdx.x;
    const int m = blockIdx.y;
    const float* t = tmp + (size_t)m * RANK;
    const float4* pr = (const float4*)(pu + (size_t)n * RANK);

    float acc = bias[n];
    #pragma unroll
    for (int r = 0; r < RANK / 4; ++r) {
        float4 p = pr[r];
        acc = fmaf(t[4 * r + 0], p.x, acc);
        acc = fmaf(t[4 * r + 1], p.y, acc);
        acc = fmaf(t[4 * r + 2], p.z, acc);
        acc = fmaf(t[4 * r + 3], p.w, acc);
    }
    out[(size_t)m * O_DIM + n] = acc;
}

__global__ __launch_bounds__(256) void gemm_w4a4_kernel(
    const int8_t* __restrict__ qact,
    const float*  __restrict__ ascales,
    const int*    __restrict__ qweight,
    const float*  __restrict__ wscales,
    float*        __restrict__ out)
{
    __shared__ alignas(16) int8_t ldsB[4 * 64 * 16];
    __shared__ alignas(16) int8_t ldsA[32 * 64];
    __shared__ alignas(16) float  ldsAsc[32 * NG];

    const int tid  = threadIdx.x;
    const int lane = tid & 31;
    const int wave = tid >> 5;
    const int wm   = wave >> 2;
    const int wn   = wave & 3;

    const int m0 = blockIdx.x * 32;
    const int n0 = blockIdx.y * 64;

    {
        const float4* src = (const float4*)(ascales + (size_t)m0 * NG);
        float4* dst = (float4*)ldsAsc;
        #pragma unroll
        for (int i = 0; i < 4; ++i)
            dst[tid + 256 * i] = src[tid + 256 * i];
    }

    float acc[8];
    #pragma unroll
    for (int k = 0; k < 8; ++k) acc[k] = 0.0f;

    const int  nLane = n0 + wn * 16 + (lane & 15);
    const int  half8 = (lane >> 4) * 8;

    const int kb = tid & 15;
    const int nb = tid >> 4;
    const int j0 = nb * 4;
    const int ns  = nb >> 2;
    const int nn0 = (nb & 3) * 4;
    int* const bdst = (int*)(ldsB + ns * 1024 + nn0);

    const int arow = tid >> 3;
    const int achk = tid & 7;

    for (int g = 0; g < NG; ++g) {
        __syncthreads();

        {
            const size_t col = (size_t)g * 64 + kb * 4;
            const int4 w0 = *(const int4*)(qweight + (size_t)(n0 + j0 + 0) * C_DIM + col);
            const int4 w1 = *(const int4*)(qweight + (size_t)(n0 + j0 + 1) * C_DIM + col);
            const int4 w2 = *(const int4*)(qweight + (size_t)(n0 + j0 + 2) * C_DIM + col);
            const int4 w3 = *(const int4*)(qweight + (size_t)(n0 + j0 + 3) * C_DIM + col);
            bdst[nn0 * 63 / 4 + 0 * 16 + kb] = pack4(w0.x - 8, w0.y - 8, w0.z - 8, w0.w - 8);
            bdst[nn0 * 63 / 4 + 1 * 16 + kb] = pack4(w1.x - 8, w1.y - 8, w1.z - 8, w1.w - 8);
            bdst[nn0 * 63 / 4 + 2 * 16 + kb] = pack4(w2.x - 8, w2.y - 8, w2.z - 8, w2.w - 8);
            bdst[nn0 * 63 / 4 + 3 * 16 + kb] = pack4(w3.x - 8, w3.y - 8, w3.z - 8, w3.w - 8);
        }
        {
            *(uint64_t*)(ldsA + arow * 64 + achk * 8) =
                *(const uint64_t*)(qact + (size_t)(m0 + arow) * C_DIM + (size_t)g * 64 + achk * 8);
        }
        __syncthreads();

        v8i a, b;
        {
            const int ar = (wm * 16 + (lane & 15)) * 64 + (lane >> 4) * 8;
            int2* ap = reinterpret_cast<int2*>(&a);
            ap[0] = *(const int2*)(ldsA + ar + 0);
            ap[1] = *(const int2*)(ldsA + ar + 16);
            ap[2] = *(const int2*)(ldsA + ar + 32);
            ap[3] = *(const int2*)(ldsA + ar + 48);
        }
        {
            const int br = wn * 1024 + (lane & 15) * 64 + (lane >> 4) * 8;
            int2* bp = reinterpret_cast<int2*>(&b);
            bp[0] = *(const int2*)(ldsB + br + 0);
            bp[1] = *(const int2*)(ldsB + br + 16);
            bp[2] = *(const int2*)(ldsB + br + 32);
            bp[3] = *(const int2*)(ldsB + br + 48);
        }

        v8i zero = {};
        v8i ps = __builtin_amdgcn_wmma_i32_16x16x64_iu8(
            true, a, true, b, zero, false, false);

        const float ws = wscales[(size_t)g * O_DIM + nLane];
        #pragma unroll
        for (int k = 0; k < 8; ++k) {
            const float as = ldsAsc[(wm * 16 + half8 + k) * NG + g];
            acc[k] = fmaf((float)ps[k], as * ws, acc[k]);
        }
    }

    const size_t ob = (size_t)(m0 + wm * 16 + half8) * O_DIM + nLane;
    #pragma unroll
    for (int k = 0; k < 8; ++k)
        out[ob + (size_t)k * O_DIM] += acc[k];
}

extern "C" void kernel_launch(void* const* d_in, const int* in_sizes, int n_in,
                              void* d_out, int out_size, void* d_ws, size_t ws_size,
                              hipStream_t stream) {
    const float* x      = (const float*)d_in[0];
    const int*   qw     = (const int*)  d_in[1];
    const float* wsc    = (const float*)d_in[2];
    const float* smooth = (const float*)d_in[3];
    const float* pd     = (const float*)d_in[4];
    const float* pu     = (const float*)d_in[5];
    const float* bias   = (const float*)d_in[6];
    float* out = (float*)d_out;

    uint8_t* ws = (uint8_t*)d_ws;
    int8_t* qact    = (int8_t*)ws;
    float*  ascales = (float*)(ws + (size_t)M_DIM * C_DIM);
    float*  tmp     = (float*)(ws + (size_t)M_DIM * C_DIM
                                  + (size_t)M_DIM * NG * sizeof(float));

    quant_act_kernel<<<dim3((M_DIM * NG) / 8), dim3(256), 0, stream>>>(
        x, smooth, qact, ascales);

    lora_down_kernel<<<dim3(M_DIM), dim3(256), 0, stream>>>(x, pd, tmp);

    lora_up_bias_kernel<<<dim3(O_DIM / 256, M_DIM), dim3(256), 0, stream>>>(
        tmp, pu, bias, out);

    gemm_w4a4_kernel<<<dim3(M_DIM / 32, O_DIM / 64), dim3(256), 0, stream>>>(
        qact, ascales, qw, wsc, out);
}
